// TransformerRadford_70403103916561
// MI455X (gfx1250) — matrix operand order repaired and checked by reading, not compiled, not run on hardware
//
#include <hip/hip_runtime.h>
#include <hip/hip_bf16.h>
#include <stdint.h>

// ---------------- problem constants ----------------
#define BB 8
#define SS 1024
#define DD 1024
#define HH 16
#define DKK 64
#define FFF 4096
#define BSR (BB * SS)          // 8192 rows

typedef __attribute__((ext_vector_type(16))) _Float16 v16h;
typedef __attribute__((ext_vector_type(8)))  float    v8f;

// ---------------- WMMA wrapper ----------------
__device__ __forceinline__ v8f wmma_f16(v16h a, v16h b, v8f c) {
  // D(f32 16x16) = A(16x32 f16) * B(32x16 f16) + C
  return __builtin_amdgcn_wmma_f32_16x16x32_f16(
      /*neg_a=*/false, a, /*neg_b=*/false, b,
      /*c_mod=*/(short)0, c, /*reuse_a=*/false, /*reuse_b=*/false);
}

// ---------------- CDNA5 async global->LDS staging ----------------
// LDS aperture rule (ISA 10.2): flat address low 32 bits == wave-relative LDS
// byte offset, which is exactly what GLOBAL_LOAD_ASYNC_TO_LDS's VDST expects.
__device__ __forceinline__ unsigned lds_off_of(const void* p) {
  return (unsigned)(uintptr_t)p;
}

// per-lane 16B async copy: LDS[lds_off] = MEM[gptr], tracked on ASYNCcnt
__device__ __forceinline__ void async_ld_b128(unsigned lds_off, const void* gptr) {
  unsigned long long ga = (unsigned long long)(uintptr_t)gptr;
  asm volatile("global_load_async_to_lds_b128 %0, %1, off"
               :: "v"(lds_off), "v"(ga) : "memory");
}

template <int N>
__device__ __forceinline__ void wait_async_le() {
#if __has_builtin(__builtin_amdgcn_s_wait_asynccnt)
  __builtin_amdgcn_s_wait_asynccnt(N);
#else
  asm volatile("s_wait_asynccnt %0" :: "i"(N) : "memory");
#endif
}

// ---------------- operand loaders (wave32; A and B both use contiguous k = 16*(lane/16) + s) ----------------
// A operand: 16x32 (MxK) f16 from row-major src (row stride ld halves).
__device__ __forceinline__ v16h load_A(const _Float16* src, int ld) {
  int lane = threadIdx.x & 31;
  int row  = lane & 15;
  int kb   = (lane < 16) ? 0 : 16;
  const _Float16* p = src + (size_t)row * ld;
  v16h a;
#pragma unroll
  for (int j = 0; j < 8; ++j) {
    a[2*j]     = p[kb + 2*j];
    a[2*j+1]   = p[kb + 2*j + 1];
  }
  return a;
}

// B operand: 32x16 (KxN) f16 from row-major [K][N] src.
__device__ __forceinline__ v16h load_B(const _Float16* src, int ld) {
  int lane = threadIdx.x & 31;
  int col  = lane & 15;
  int kb   = (lane < 16) ? 0 : 16;
  v16h b;
#pragma unroll
  for (int j = 0; j < 8; ++j) {
    b[2*j]   = src[(size_t)(kb + 2*j) * ld + col];
    b[2*j+1] = src[(size_t)(kb + 2*j + 1) * ld + col];
  }
  return b;
}

// B operand from row-major [N][K] source (NT GEMM): B[k][n] = src[n][k].
__device__ __forceinline__ v16h load_B_nt(const _Float16* src, int ld) {
  int lane = threadIdx.x & 31;
  int n    = lane & 15;
  int kb   = (lane < 16) ? 0 : 16;
  const _Float16* p = src + (size_t)n * ld + kb;
  v16h b;
#pragma unroll
  for (int j = 0; j < 8; ++j) {
    b[2*j]   = p[2*j];
    b[2*j+1] = p[2*j + 1];
  }
  return b;
}

// ---------------- f32 -> f16 convert ----------------
__global__ void conv_f16_kernel(const float* __restrict__ src,
                                _Float16* __restrict__ dst, int n4) {
  int i = blockIdx.x * blockDim.x + threadIdx.x;
  if (i < n4) {
    float4 v = reinterpret_cast<const float4*>(src)[i];
    __align__(8) _Float16 h[4] = {(_Float16)v.x, (_Float16)v.y,
                                  (_Float16)v.z, (_Float16)v.w};
    reinterpret_cast<float2*>(dst)[i] = *reinterpret_cast<float2*>(h);
  }
}

// ---------------- generic WMMA GEMM, double-buffered async pipeline ---------
// C[M,N] = A[M,K] * B + bias ; BT=1: B row-major [N,K] (NT), BT=0: [K,N] (NN)
// 256 threads = 8 waves; block tile 128(M) x 64(N) x 64(K); wave tile 32x32.
// Tiles staged with GLOBAL_LOAD_ASYNC_TO_LDS_B128; every thread issues exactly
// 6 async ops per k-tile (4 for A, 2 for B) so `s_wait_asynccnt 6` after
// issuing tile i+1 guarantees tile i is resident (per-wave counter, in-order).
template <int BT, int RELU, int OUT16>
__global__ void gemm_wmma_kernel(const _Float16* __restrict__ A,
                                 const _Float16* __restrict__ Bm,
                                 const float* __restrict__ bias,
                                 float* __restrict__ Cf,
                                 _Float16* __restrict__ Ch,
                                 int M, int N, int K, int lda, int ldb, int ldc,
                                 long Abs, long Bbs, long Cbs) {
  A  += (size_t)blockIdx.z * (size_t)Abs;
  Bm += (size_t)blockIdx.z * (size_t)Bbs;
  if (OUT16) Ch += (size_t)blockIdx.z * (size_t)Cbs;
  else       Cf += (size_t)blockIdx.z * (size_t)Cbs;

  const int m0 = blockIdx.y * 128;
  const int n0 = blockIdx.x * 64;
  const int t = threadIdx.x;
  const int wid = t >> 5, lane = t & 31;
  const int col = lane & 15, rbase = (lane < 16) ? 0 : 8;
  const int wm = wid & 3, wn = wid >> 2;

  __shared__ _Float16 As[2][128 * 72];  // 128 rows x 64 k, pitch 72
  __shared__ _Float16 Bs[2][64 * 72];   // 64 rows x 64, pitch 72 (NT and NN)

  v8f c00 = {}, c01 = {}, c10 = {}, c11 = {};

  // A staging: 2 threads/row, 32 halves each (4 x b128)
  const int arow = t >> 1, aseg = (t & 1) * 32;
  const unsigned as_lo = lds_off_of(&As[0][arow * 72 + aseg]);
  // B staging: 4 threads/row, 16 halves each (2 x b128)
  const int brow = t >> 2, bseg = (t & 3) * 16;
  const unsigned bs_lo = lds_off_of(&Bs[0][brow * 72 + bseg]);
  const unsigned AS_STRIDE = 128u * 72u * 2u;  // bytes per A buffer
  const unsigned BS_STRIDE = 64u * 72u * 2u;   // bytes per B buffer

  auto stage = [&](int k0, int buf) {
    const _Float16* ga = A + (size_t)(m0 + arow) * lda + k0 + aseg;
    unsigned alo = as_lo + (unsigned)buf * AS_STRIDE;
#pragma unroll
    for (int j = 0; j < 4; ++j) async_ld_b128(alo + 16u * j, ga + 8 * j);
    const _Float16* gb = BT
        ? (Bm + (size_t)(n0 + brow) * ldb + k0 + bseg)   // [N,K]
        : (Bm + (size_t)(k0 + brow) * ldb + n0 + bseg);  // [K,N]
    unsigned blo = bs_lo + (unsigned)buf * BS_STRIDE;
    async_ld_b128(blo, gb);
    async_ld_b128(blo + 16u, gb + 8);
  };

  const int nk = K >> 6;
  stage(0, 0);
  for (int ki = 0; ki < nk; ++ki) {
    const int cur = ki & 1;
    if (ki + 1 < nk) {
      stage((ki + 1) << 6, cur ^ 1);
      wait_async_le<6>();   // tile ki resident (6 newer ops may remain in flight)
    } else {
      wait_async_le<0>();
    }
    __syncthreads();

#pragma unroll
    for (int kk = 0; kk < 64; kk += 32) {
      v16h a0 = load_A(&As[cur][(wm * 32) * 72 + kk], 72);
      v16h a1 = load_A(&As[cur][(wm * 32 + 16) * 72 + kk], 72);
      v16h b0, b1;
      if (BT) {
        b0 = load_B_nt(&Bs[cur][(wn * 32) * 72 + kk], 72);
        b1 = load_B_nt(&Bs[cur][(wn * 32 + 16) * 72 + kk], 72);
      } else {
        b0 = load_B(&Bs[cur][kk * 72 + wn * 32], 72);
        b1 = load_B(&Bs[cur][kk * 72 + wn * 32 + 16], 72);
      }
      c00 = wmma_f16(a0, b0, c00);
      c01 = wmma_f16(a0, b1, c01);
      c10 = wmma_f16(a1, b0, c10);
      c11 = wmma_f16(a1, b1, c11);
    }
    __syncthreads();  // reads done before this buffer is refilled (iter ki+2)
  }

  // epilogue
  const int mrow = m0 + wm * 32;
  const int ncol = n0 + wn * 32;
#pragma unroll
  for (int i = 0; i < 2; ++i) {
#pragma unroll
    for (int j = 0; j < 2; ++j) {
      const v8f& c = (i == 0) ? ((j == 0) ? c00 : c01) : ((j == 0) ? c10 : c11);
      int cg = ncol + j * 16 + col;
      float bv = (bias != nullptr) ? bias[cg] : 0.0f;
#pragma unroll
      for (int r = 0; r < 8; ++r) {
        float v = c[r] + bv;
        if (RELU) v = fmaxf(v, 0.0f);
        size_t idx = (size_t)(mrow + i * 16 + rbase + r) * ldc + cg;
        if (OUT16) Ch[idx] = (_Float16)v;
        else       Cf[idx] = v;
      }
    }
  }
}

// ---------------- attention pass 1: per-key column softmax stats ----------------
// Reference quirk: s = where(mask, QK^T, -1000) / sqrt(dk); softmax over the
// QUERY axis -> need max/sum over q for each key column.
__global__ void attn_colstats_kernel(const _Float16* __restrict__ qh,
                                     const _Float16* __restrict__ kh,
                                     const unsigned char* __restrict__ mask,
                                     float* __restrict__ mstat,
                                     float* __restrict__ lstat) {
  const int bh = blockIdx.y, b = bh >> 4, h = bh & 15;
  const int kblk = blockIdx.x;
  const int t = threadIdx.x, wid = t >> 5, lane = t & 31;
  const int col = lane & 15, rbase = (lane < 16) ? 0 : 8;

  const _Float16* Qb = qh + (size_t)(b * SS) * DD + h * DKK;
  const _Float16* Kb = kh + (size_t)(b * SS) * DD + h * DKK;
  const unsigned char* mb = mask + (size_t)b * SS * SS;

  float m_run[4], l_run[4];
#pragma unroll
  for (int i = 0; i < 4; ++i) { m_run[i] = -3.0e38f; l_run[i] = 0.0f; }

  // K-block operands reused across all q tiles: hoist.
  v16h kb0[4], kb1[4];
#pragma unroll
  for (int kg = 0; kg < 4; ++kg) {
    int k0 = kblk * 64 + kg * 16;
    kb0[kg] = load_B_nt(Kb + (size_t)k0 * DD, DD);
    kb1[kg] = load_B_nt(Kb + (size_t)k0 * DD + 32, DD);
  }

  for (int it = 0; it < 8; ++it) {
    int q0 = it * 128 + wid * 16;
    v16h a0 = load_A(Qb + (size_t)q0 * DD, DD);
    v16h a1 = load_A(Qb + (size_t)q0 * DD + 32, DD);
#pragma unroll
    for (int kg = 0; kg < 4; ++kg) {
      int k0 = kblk * 64 + kg * 16;
      v8f c = {};
      c = wmma_f16(a0, kb0[kg], c);
      c = wmma_f16(a1, kb1[kg], c);
      float sv[8];
      float tm = -3.0e38f;
#pragma unroll
      for (int r = 0; r < 8; ++r) {
        int q = q0 + rbase + r;
        float s = mb[(size_t)q * SS + (k0 + col)] ? c[r] : -1000.0f;
        s *= 0.125f;  // 1/sqrt(64)
        sv[r] = s;
        tm = fmaxf(tm, s);
      }
      tm = fmaxf(tm, __shfl_xor(tm, 16, 32));  // merge rows 0-7 with 8-15
      float tl = 0.0f;
#pragma unroll
      for (int r = 0; r < 8; ++r) tl += __expf(sv[r] - tm);
      tl += __shfl_xor(tl, 16, 32);
      float mo = fmaxf(m_run[kg], tm);
      l_run[kg] = l_run[kg] * __expf(m_run[kg] - mo) + tl * __expf(tm - mo);
      m_run[kg] = mo;
    }
  }

  __shared__ float pm[8][64], pl[8][64];
  if (lane < 16) {
#pragma unroll
    for (int kg = 0; kg < 4; ++kg) {
      pm[wid][kg * 16 + col] = m_run[kg];
      pl[wid][kg * 16 + col] = l_run[kg];
    }
  }
  __syncthreads();
  if (t < 64) {
    float m = -3.0e38f;
#pragma unroll
    for (int w = 0; w < 8; ++w) m = fmaxf(m, pm[w][t]);
    float l = 0.0f;
#pragma unroll
    for (int w = 0; w < 8; ++w) l += pl[w][t] * __expf(pm[w][t] - m);
    mstat[(size_t)bh * SS + kblk * 64 + t] = m;
    lstat[(size_t)bh * SS + kblk * 64 + t] = l;
  }
}

// ---------------- attention pass 2: O = P V (P normalized per column) --------
__global__ void attn_pv_kernel(const _Float16* __restrict__ qh,
                               const _Float16* __restrict__ kh,
                               const _Float16* __restrict__ vh,
                               const unsigned char* __restrict__ mask,
                               const float* __restrict__ mstat,
                               const float* __restrict__ lstat,
                               _Float16* __restrict__ oh) {
  const int bh = blockIdx.y, b = bh >> 4, h = bh & 15;
  const int q0b = blockIdx.x * 64;
  const int t = threadIdx.x, wid = t >> 5, lane = t & 31;
  const int col = lane & 15, rbase = (lane < 16) ? 0 : 8;
  const int qg = wid >> 1, eg = wid & 1;

  const _Float16* Qb = qh + (size_t)(b * SS) * DD + h * DKK;
  const _Float16* Kb = kh + (size_t)(b * SS) * DD + h * DKK;
  const _Float16* Vb = vh + (size_t)(b * SS) * DD + h * DKK;
  const unsigned char* mb = mask + (size_t)b * SS * SS;

  __shared__ _Float16 Q_lds[64 * 72];
  __shared__ _Float16 P_lds[64 * 72];

  {  // stage the 64x64 Q block once via async copy
    int row = t >> 2, seg = (t & 3) * 16;
    const _Float16* gq = Qb + (size_t)(q0b + row) * DD + seg;
    unsigned lo = lds_off_of(&Q_lds[row * 72 + seg]);
    async_ld_b128(lo, gq);
    async_ld_b128(lo + 16, gq + 8);
  }
  wait_async_le<0>();
  __syncthreads();

  v16h a0 = load_A(&Q_lds[(qg * 16) * 72], 72);
  v16h a1 = load_A(&Q_lds[(qg * 16) * 72 + 32], 72);

  v8f c0 = {}, c1 = {};

  for (int kbi = 0; kbi < 16; ++kbi) {
    int k0 = kbi * 64;
    // scores -> probabilities for this wave's 2 key sub-tiles
#pragma unroll
    for (int kgi = 0; kgi < 2; ++kgi) {
      int kg = eg * 2 + kgi;
      int kk0 = k0 + kg * 16;
      v16h b0 = load_B_nt(Kb + (size_t)kk0 * DD, DD);
      v16h b1 = load_B_nt(Kb + (size_t)kk0 * DD + 32, DD);
      v8f s = {};
      s = wmma_f16(a0, b0, s);
      s = wmma_f16(a1, b1, s);
      int kcol = kk0 + col;
      float mk = mstat[(size_t)bh * SS + kcol];
      float rlk = 1.0f / lstat[(size_t)bh * SS + kcol];
#pragma unroll
      for (int r = 0; r < 8; ++r) {
        int q = q0b + qg * 16 + rbase + r;
        float sc = mb[(size_t)q * SS + kcol] ? s[r] : -1000.0f;
        float p = __expf(sc * 0.125f - mk) * rlk;
        P_lds[(qg * 16 + rbase + r) * 72 + kg * 16 + col] = (_Float16)p;
      }
    }
    __syncthreads();
    // O += P * V
#pragma unroll
    for (int kk = 0; kk < 2; ++kk) {
      v16h pa = load_A(&P_lds[(qg * 16) * 72 + kk * 32], 72);
      v16h vb0 = load_B(Vb + (size_t)(k0 + kk * 32) * DD + eg * 32, DD);
      v16h vb1 = load_B(Vb + (size_t)(k0 + kk * 32) * DD + eg * 32 + 16, DD);
      c0 = wmma_f16(pa, vb0, c0);
      c1 = wmma_f16(pa, vb1, c1);
    }
    __syncthreads();
  }

#pragma unroll
  for (int r = 0; r < 8; ++r) {
    int q = q0b + qg * 16 + rbase + r;
    size_t ro = (size_t)(b * SS + q) * DD + h * DKK + eg * 32;
    oh[ro + col]      = (_Float16)c0[r];
    oh[ro + 16 + col] = (_Float16)c1[r];
  }
}

// ---------------- residual add + LayerNorm (f32 out + optional f16 out) ------
__global__ void add_ln_kernel(const float* __restrict__ a,
                              const float* __restrict__ res,
                              const float* __restrict__ g,
                              const float* __restrict__ be,
                              float* __restrict__ outf,
                              _Float16* __restrict__ outh) {
  const int row = blockIdx.x;
  const int t = threadIdx.x;
  const float* ar = a + (size_t)row * DD;
  const float* rr = res + (size_t)row * DD;
  float v[4], sum = 0.0f, sq = 0.0f;
#pragma unroll
  for (int i = 0; i < 4; ++i) {
    int c = i * 256 + t;
    v[i] = ar[c] + rr[c];
    sum += v[i];
    sq += v[i] * v[i];
  }
  __shared__ float r1[256], r2[256];
  r1[t] = sum; r2[t] = sq;
  __syncthreads();
  for (int s = 128; s > 0; s >>= 1) {
    if (t < s) { r1[t] += r1[t + s]; r2[t] += r2[t + s]; }
    __syncthreads();
  }
  const float mu = r1[0] * (1.0f / DD);
  const float var = r2[0] * (1.0f / DD) - mu * mu;
  const float rs = rsqrtf(var + 1e-5f);
#pragma unroll
  for (int i = 0; i < 4; ++i) {
    int c = i * 256 + t;
    float y = (v[i] - mu) * rs * g[c] + be[c];
    outf[(size_t)row * DD + c] = y;
    if (outh) outh[(size_t)row * DD + c] = (_Float16)y;
  }
}

// ---------------- host launcher ----------------
extern "C" void kernel_launch(void* const* d_in, const int* in_sizes, int n_in,
                              void* d_out, int out_size, void* d_ws, size_t ws_size,
                              hipStream_t stream) {
  const float* x      = (const float*)d_in[0];
  const unsigned char* mask = (const unsigned char*)d_in[1];
  const float* wq     = (const float*)d_in[2];
  const float* wk     = (const float*)d_in[3];
  const float* wv     = (const float*)d_in[4];
  const float* fc_w   = (const float*)d_in[5];
  const float* fc_b   = (const float*)d_in[6];
  const float* w1     = (const float*)d_in[7];
  const float* b1     = (const float*)d_in[8];
  const float* w2     = (const float*)d_in[9];
  const float* b2     = (const float*)d_in[10];
  const float* ln1_g  = (const float*)d_in[11];
  const float* ln1_b  = (const float*)d_in[12];
  const float* ln2_g  = (const float*)d_in[13];
  const float* ln2_b  = (const float*)d_in[14];
  float* out = (float*)d_out;

  // workspace carve-out (256B aligned)
  char* wsp = (char*)d_ws;
  size_t off = 0;
  auto alloc = [&](size_t bytes) -> void* {
    off = (off + 255) & ~(size_t)255;
    void* p = wsp + off;
    off += bytes;
    return p;
  };
  _Float16* xh  = (_Float16*)alloc((size_t)BSR * DD * 2);
  _Float16* wqh = (_Float16*)alloc((size_t)HH * DD * DKK * 2);
  _Float16* wkh = (_Float16*)alloc((size_t)HH * DD * DKK * 2);
  _Float16* wvh = (_Float16*)alloc((size_t)HH * DD * DKK * 2);
  _Float16* fch = (_Float16*)alloc((size_t)DD * DD * 2);
  _Float16* w1h = (_Float16*)alloc((size_t)FFF * DD * 2);
  _Float16* w2h = (_Float16*)alloc((size_t)DD * FFF * 2);
  // big region: q/k/v/o (4 x 16MB) later reused as h1 (8192x4096 f16 = 64MB)
  _Float16* qh = (_Float16*)alloc((size_t)4 * BSR * DD * 2);
  _Float16* kh = qh + (size_t)BSR * DD;
  _Float16* vh = kh + (size_t)BSR * DD;
  _Float16* oh = vh + (size_t)BSR * DD;
  _Float16* h1 = qh;  // reuse after attention completes
  float* att  = (float*)alloc((size_t)BSR * DD * 4);  // also reused for h2
  float* x1f  = (float*)alloc((size_t)BSR * DD * 4);
  _Float16* x1h = (_Float16*)alloc((size_t)BSR * DD * 2);
  float* mstat = (float*)alloc((size_t)BB * HH * SS * 4);
  float* lstat = (float*)alloc((size_t)BB * HH * SS * 4);
  float* h2 = att;
  (void)in_sizes; (void)n_in; (void)out_size; (void)ws_size;

  // 1) f32 -> f16 conversions
  auto conv = [&](const float* s, _Float16* d, size_t n) {
    int n4 = (int)(n / 4);
    conv_f16_kernel<<<(n4 + 255) / 256, 256, 0, stream>>>(s, d, n4);
  };
  conv(x, xh, (size_t)BSR * DD);
  conv(wq, wqh, (size_t)HH * DD * DKK);
  conv(wk, wkh, (size_t)HH * DD * DKK);
  conv(wv, wvh, (size_t)HH * DD * DKK);
  conv(fc_w, fch, (size_t)DD * DD);
  conv(w1, w1h, (size_t)FFF * DD);
  conv(w2, w2h, (size_t)DD * FFF);

  // 2) QKV projections: per-head NN GEMM (wq[h] is [D, DK] = [K,N] row-major)
  {
    dim3 grid(1, BSR / 128, HH);
    gemm_wmma_kernel<0, 0, 1><<<grid, 256, 0, stream>>>(
        xh, wqh, nullptr, nullptr, qh, BSR, DKK, DD, DD, DKK, DD,
        0L, (long)DD * DKK, (long)DKK);
    gemm_wmma_kernel<0, 0, 1><<<grid, 256, 0, stream>>>(
        xh, wkh, nullptr, nullptr, kh, BSR, DKK, DD, DD, DKK, DD,
        0L, (long)DD * DKK, (long)DKK);
    gemm_wmma_kernel<0, 0, 1><<<grid, 256, 0, stream>>>(
        xh, wvh, nullptr, nullptr, vh, BSR, DKK, DD, DD, DKK, DD,
        0L, (long)DD * DKK, (long)DKK);
  }

  // 3) attention: column-softmax stats, then P*V
  {
    dim3 grid(SS / 64, BB * HH);
    attn_colstats_kernel<<<grid, 256, 0, stream>>>(qh, kh, mask, mstat, lstat);
    attn_pv_kernel<<<grid, 256, 0, stream>>>(qh, kh, vh, mask, mstat, lstat, oh);
  }

  // 4) output projection: att = oh @ fc_w^T + fc_b  (NT, f32 out)
  {
    dim3 grid(DD / 64, BSR / 128, 1);
    gemm_wmma_kernel<1, 0, 0><<<grid, 256, 0, stream>>>(
        oh, fch, fc_b, att, nullptr, BSR, DD, DD, DD, DD, DD, 0L, 0L, 0L);
  }

  // 5) x1 = LN(att + x)  (f32 + f16 copies)
  add_ln_kernel<<<BSR, 256, 0, stream>>>(att, x, ln1_g, ln1_b, x1f, x1h);

  // 6) h1 = relu(x1 @ w1^T + b1)  (NT, f16 out) -- reuses q/k/v/o region
  {
    dim3 grid(FFF / 64, BSR / 128, 1);
    gemm_wmma_kernel<1, 1, 1><<<grid, 256, 0, stream>>>(
        x1h, w1h, b1, nullptr, h1, BSR, FFF, DD, DD, DD, FFF, 0L, 0L, 0L);
  }

  // 7) h2 = h1 @ w2^T + b2  (NT, f32 out) -- reuses att buffer
  {
    dim3 grid(DD / 64, BSR / 128, 1);
    gemm_wmma_kernel<1, 0, 0><<<grid, 256, 0, stream>>>(
        h1, w2h, b2, h2, nullptr, BSR, DD, FFF, FFF, FFF, DD, 0L, 0L, 0L);
  }

  // 8) out = LN(h2 + x1)
  add_ln_kernel<<<BSR, 256, 0, stream>>>(h2, x1f, ln2_g, ln2_b, out, nullptr);
}
